// Attention_1984274890902
// MI455X (gfx1250) — matrix operand order repaired and checked by reading, not compiled, not run on hardware
//
#include <hip/hip_runtime.h>
#include <hip/hip_bf16.h>
#include <stdint.h>

// ---------------------------------------------------------------------------
// MI455X (gfx1250) fused attention pipeline, bf16 WMMA everywhere.
// ---------------------------------------------------------------------------

typedef unsigned short u16;
typedef __attribute__((ext_vector_type(4)))  unsigned int u32x4;
typedef __attribute__((ext_vector_type(16))) __bf16       v16bf;
typedef __attribute__((ext_vector_type(8)))  float        v8f;

static constexpr int BB  = 8;
static constexpr int NN  = 1280;    // tokens per batch (256 merged + 1024 spatial)
static constexpr int CC  = 768;
static constexpr int HH  = 12;
static constexpr int HD  = 64;
static constexpr int NT  = 256;     // merged tokens
static constexpr int C3  = 3 * CC;  // 2304
static constexpr int MT  = BB * NN; // 10240 total token rows

__device__ __forceinline__ u16 f2bf(float f) {
  union { float f; unsigned u; } x{f};
  unsigned u = x.u;
  u += 0x7fffu + ((u >> 16) & 1u);   // round-to-nearest-even
  return (u16)(u >> 16);
}

__device__ __forceinline__ v16bf make_frag(u32x4 a, u32x4 b) {
  union { u32x4 q[2]; v16bf v; } u;
  u.q[0] = a; u.q[1] = b;
  return u.v;
}

// A-matrix fragment (16x32, MxK): lane holds row m=lane&15; K chunks at
// kb = (lane>=16 ? 8 : 0) and kb+16, each 8 contiguous bf16 (16 bytes).
__device__ __forceinline__ v16bf loadA(const u16* rowp, int lane) {
  int kb = (lane >> 4) << 3;
  u32x4 a = *(const u32x4*)(rowp + kb);
  u32x4 b = *(const u32x4*)(rowp + kb + 16);
  return make_frag(a, b);
}

// B-matrix fragment (32x16, KxN): lane holds column n=lane&15; K chunks at
// kb = (lane>=16 ? 8 : 0) and kb+16, each 8 contiguous bf16 (16 bytes).
__device__ __forceinline__ v16bf loadB(const u16* colp, int lane) {
  int kb = (lane >> 4) << 3;
  u32x4 a = *(const u32x4*)(colp + kb);
  u32x4 b = *(const u32x4*)(colp + kb + 16);
  return make_frag(a, b);
}

__device__ __forceinline__ v8f wmma_bf16(v16bf a, v16bf b, v8f c) {
  return __builtin_amdgcn_wmma_f32_16x16x32_bf16(
      false, a, false, b, (short)0, c, false, false);
}

// ---------------------------------------------------------------------------
// Kernel 0: fp32 -> bf16 conversion.
// ---------------------------------------------------------------------------
__global__ void k_cvt(const float* __restrict__ s, u16* __restrict__ d, int n) {
  for (int i = blockIdx.x * blockDim.x + threadIdx.x; i < n;
       i += gridDim.x * blockDim.x)
    d[i] = f2bf(s[i]);
}

// ---------------------------------------------------------------------------
// Kernel 1: QKV GEMM  [10240 x 768] x [768 x 2304]  (x @ qkv_w^T)
// 64x64 tile, 4 waves (2x2 of 32x32), K step 32, double-buffered LDS.
// Scatters Q (scaled by 1/8), K, and V transposed ([hd][N] per head).
// All scatter decode is block-uniform (tiles never straddle batch or q/k/v
// boundaries) -> pure scalar decode, no per-lane division.
// ---------------------------------------------------------------------------
__global__ void __launch_bounds__(128)
k_qkv(const u16* __restrict__ xbf, const u16* __restrict__ wq,
      u16* __restrict__ Q, u16* __restrict__ K, u16* __restrict__ Vt) {
  __shared__ __align__(16) u16 As[2][64 * 40];   // 32 cols + 8 pad
  __shared__ __align__(16) u16 Bs[2][64 * 40];
  const int tid = threadIdx.x, lane = tid & 31, w = tid >> 5;
  const int wm = (w >> 1) * 32, wn = (w & 1) * 32;
  const int m0 = (int)blockIdx.x * 64;
  const int n0 = (int)blockIdx.y * 64;
  const int n = lane & 15, hi = lane >> 4;

  auto load_tile = [&](int k0, int buf) {
#pragma unroll
    for (int i = 0; i < 2; ++i) {            // 128 thr x 2 x 16B per matrix
      int chunk = tid * 2 + i;
      int r = chunk >> 2, cc = chunk & 3;
      *(u32x4*)&As[buf][r * 40 + cc * 8] =
          *(const u32x4*)&xbf[(m0 + r) * CC + k0 + cc * 8];
      *(u32x4*)&Bs[buf][r * 40 + cc * 8] =
          *(const u32x4*)&wq[(n0 + r) * CC + k0 + cc * 8];
    }
  };

  v8f acc[2][2] = {};
  load_tile(0, 0);
  __syncthreads();
  int cur = 0;
  for (int k0 = 0; k0 < CC; k0 += 32) {
    if (k0 + 32 < CC) load_tile(k0 + 32, cur ^ 1);
    v16bf a[2], b[2];
#pragma unroll
    for (int i = 0; i < 2; ++i)
      a[i] = loadA(&As[cur][(wm + i * 16 + n) * 40], lane);
#pragma unroll
    for (int j = 0; j < 2; ++j)
      b[j] = loadB(&Bs[cur][(wn + j * 16 + n) * 40], lane);
#pragma unroll
    for (int i = 0; i < 2; ++i)
#pragma unroll
      for (int j = 0; j < 2; ++j) acc[i][j] = wmma_bf16(a[i], b[j], acc[i][j]);
    __syncthreads();
    cur ^= 1;
  }

  // Block-uniform decode (scalar).
  const int b_    = m0 / NN;            // batch
  const int nbase = m0 - b_ * NN;       // token base within batch
  const int t     = n0 / CC;            // 0=q, 1=k, 2=v
  const int cb    = n0 - t * CC;        // column base within 768
  const int hb    = b_ * HH;

  if (t == 0) {
#pragma unroll
    for (int i = 0; i < 2; ++i)
#pragma unroll
      for (int j = 0; j < 2; ++j) {
        int col = cb + wn + j * 16 + n;
        int h = col >> 6, d = col & 63;
        int base = ((hb + h) * NN + nbase + wm + i * 16 + hi * 8) * HD + d;
#pragma unroll
        for (int v = 0; v < 8; ++v)
          Q[base + v * HD] = f2bf(acc[i][j][v] * 0.125f);
      }
  } else if (t == 1) {
#pragma unroll
    for (int i = 0; i < 2; ++i)
#pragma unroll
      for (int j = 0; j < 2; ++j) {
        int col = cb + wn + j * 16 + n;
        int h = col >> 6, d = col & 63;
        int base = ((hb + h) * NN + nbase + wm + i * 16 + hi * 8) * HD + d;
#pragma unroll
        for (int v = 0; v < 8; ++v)
          K[base + v * HD] = f2bf(acc[i][j][v]);
      }
  } else {
#pragma unroll
    for (int i = 0; i < 2; ++i)
#pragma unroll
      for (int j = 0; j < 2; ++j) {
        int col = cb + wn + j * 16 + n;
        int h = col >> 6, d = col & 63;
        int base = ((hb + h) * HD + d) * NN + nbase + wm + i * 16 + hi * 8;
#pragma unroll
        for (int v = 0; v < 8; ++v)
          Vt[base + v] = f2bf(acc[i][j][v]);
      }
  }
}

// ---------------------------------------------------------------------------
// Kernel 2: merged-token attention. Per wave: 16 queries x 256 keys x 64 d.
// Scores in 16 f32 accum frags; width-16 shfl softmax; P bounced through LDS
// to re-layout C-frag -> A-frag for the P@V WMMAs.
// ---------------------------------------------------------------------------
__global__ void __launch_bounds__(128)
k_attn_t(const u16* __restrict__ Q, const u16* __restrict__ K,
         const u16* __restrict__ Vt, u16* __restrict__ Y) {
  __shared__ __align__(16) u16 Ps[4][16 * 264];  // 256 keys + 8 pad per row
  const int tid = threadIdx.x, lane = tid & 31, w = tid >> 5;
  const int bh = (int)blockIdx.x >> 2, qt = (int)blockIdx.x & 3;
  const int qkb = bh * NN * HD;   // Q/K base for (b,h)
  const int vb  = bh * HD * NN;   // Vt base
  const int q0 = qt * 64 + w * 16;
  const int n = lane & 15, hi = lane >> 4;

  v16bf qa[2];
#pragma unroll
  for (int i = 0; i < 2; ++i)
    qa[i] = loadA(&Q[qkb + (q0 + n) * HD + i * 32], lane);

  v8f s[16];
#pragma unroll
  for (int kt = 0; kt < 16; ++kt) {
    v16bf b0 = loadB(&K[qkb + (kt * 16 + n) * HD + 0], lane);
    v16bf b1 = loadB(&K[qkb + (kt * 16 + n) * HD + 32], lane);
    v8f t = {};
    t = wmma_bf16(qa[0], b0, t);
    t = wmma_bf16(qa[1], b1, t);
    s[kt] = t;
  }

  float mx[8], sm[8];
#pragma unroll
  for (int v = 0; v < 8; ++v) {
    float m = -1e30f;
#pragma unroll
    for (int kt = 0; kt < 16; ++kt) m = fmaxf(m, s[kt][v]);
#pragma unroll
    for (int off = 8; off; off >>= 1) m = fmaxf(m, __shfl_xor(m, off, 16));
    mx[v] = m;
  }
#pragma unroll
  for (int v = 0; v < 8; ++v) {
    float a = 0.f;
#pragma unroll
    for (int kt = 0; kt < 16; ++kt) {
      float p = __expf(s[kt][v] - mx[v]);
      s[kt][v] = p;
      a += p;
    }
#pragma unroll
    for (int off = 8; off; off >>= 1) a += __shfl_xor(a, off, 16);
    sm[v] = a;
  }

#pragma unroll
  for (int kt = 0; kt < 16; ++kt)
#pragma unroll
    for (int v = 0; v < 8; ++v)
      Ps[w][(hi * 8 + v) * 264 + kt * 16 + n] = f2bf(s[kt][v]);
  asm volatile("s_wait_dscnt 0" ::: "memory");

  v8f o[4];
#pragma unroll
  for (int g = 0; g < 4; ++g) {
    v8f t = {};
#pragma unroll
    for (int k8 = 0; k8 < 8; ++k8) {
      v16bf pa = loadA(&Ps[w][n * 264 + k8 * 32], lane);
      v16bf vv = loadB(&Vt[vb + (g * 16 + n) * NN + k8 * 32], lane);
      t = wmma_bf16(pa, vv, t);
    }
    o[g] = t;
  }

  const int b_ = bh / HH, h = bh - b_ * HH;   // uniform scalar decode
#pragma unroll
  for (int g = 0; g < 4; ++g)
#pragma unroll
    for (int v = 0; v < 8; ++v) {
      int tok = q0 + hi * 8 + v;   // merged tokens occupy global rows [0,256)
      Y[(b_ * NN + tok) * CC + h * HD + g * 16 + n] = f2bf(o[g][v] / sm[v]);
    }
}

// ---------------------------------------------------------------------------
// Kernel 3: spatial attention, flash-style online softmax over 1280 keys
// in 32-key steps (scores don't fit in registers).
// ---------------------------------------------------------------------------
__global__ void __launch_bounds__(128)
k_attn_s(const u16* __restrict__ Q, const u16* __restrict__ K,
         const u16* __restrict__ Vt, u16* __restrict__ Y) {
  __shared__ __align__(16) u16 Ps[4][16 * 40];   // 32 keys + 8 pad per row
  const int tid = threadIdx.x, lane = tid & 31, w = tid >> 5;
  const int bh = (int)blockIdx.x >> 4, qt = (int)blockIdx.x & 15;
  const int qkb = bh * NN * HD;
  const int vb  = qkb;
  const int q0 = NT + qt * 64 + w * 16;   // spatial queries start at token 256
  const int n = lane & 15, hi = lane >> 4;

  v16bf qa[2];
#pragma unroll
  for (int i = 0; i < 2; ++i)
    qa[i] = loadA(&Q[qkb + (q0 + n) * HD + i * 32], lane);

  float mrun[8], lrun[8];
  v8f o[4];
  v8f zero = {};
#pragma unroll
  for (int v = 0; v < 8; ++v) { mrun[v] = -1e30f; lrun[v] = 0.f; }
#pragma unroll
  for (int g = 0; g < 4; ++g) o[g] = zero;

  for (int kt = 0; kt < 40; ++kt) {        // 40 * 32 = 1280 keys
    const int key0 = kt * 32;
    v8f s0 = {}, s1 = {};
    {
      v16bf b00 = loadB(&K[qkb + (key0 + n) * HD + 0], lane);
      v16bf b01 = loadB(&K[qkb + (key0 + n) * HD + 32], lane);
      s0 = wmma_bf16(qa[0], b00, s0);
      s0 = wmma_bf16(qa[1], b01, s0);
      v16bf b10 = loadB(&K[qkb + (key0 + 16 + n) * HD + 0], lane);
      v16bf b11 = loadB(&K[qkb + (key0 + 16 + n) * HD + 32], lane);
      s1 = wmma_bf16(qa[0], b10, s1);
      s1 = wmma_bf16(qa[1], b11, s1);
    }

    float alpha[8];
#pragma unroll
    for (int v = 0; v < 8; ++v) {
      float m = fmaxf(s0[v], s1[v]);
#pragma unroll
      for (int off = 8; off; off >>= 1) m = fmaxf(m, __shfl_xor(m, off, 16));
      float mnew = fmaxf(mrun[v], m);
      alpha[v] = __expf(mrun[v] - mnew);
      float p0 = __expf(s0[v] - mnew);
      float p1 = __expf(s1[v] - mnew);
      s0[v] = p0; s1[v] = p1;
      float r = p0 + p1;
#pragma unroll
      for (int off = 8; off; off >>= 1) r += __shfl_xor(r, off, 16);
      lrun[v] = lrun[v] * alpha[v] + r;
      mrun[v] = mnew;
    }

#pragma unroll
    for (int v = 0; v < 8; ++v) {
      Ps[w][(hi * 8 + v) * 40 + n]      = f2bf(s0[v]);
      Ps[w][(hi * 8 + v) * 40 + 16 + n] = f2bf(s1[v]);
    }
    asm volatile("s_wait_dscnt 0" ::: "memory");

    v16bf pa = loadA(&Ps[w][n * 40], lane);
#pragma unroll
    for (int g = 0; g < 4; ++g) {
      v16bf vv = loadB(&Vt[vb + (g * 16 + n) * NN + key0], lane);
      v8f t = o[g];
#pragma unroll
      for (int v = 0; v < 8; ++v) t[v] *= alpha[v];
      o[g] = wmma_bf16(pa, vv, t);
    }
  }

  const int b_ = bh / HH, h = bh - b_ * HH;
#pragma unroll
  for (int g = 0; g < 4; ++g)
#pragma unroll
    for (int v = 0; v < 8; ++v) {
      int tok = q0 + hi * 8 + v;
      Y[(b_ * NN + tok) * CC + h * HD + g * 16 + n] = f2bf(o[g][v] / lrun[v]);
    }
}

// ---------------------------------------------------------------------------
// Kernel 4: output projection  [10240 x 768] x [768 x 768] + bias -> f32 out.
// Same double-buffered tiling as k_qkv.
// ---------------------------------------------------------------------------
__global__ void __launch_bounds__(128)
k_proj(const u16* __restrict__ Y, const u16* __restrict__ wp,
       const float* __restrict__ bias, float* __restrict__ out) {
  __shared__ __align__(16) u16 As[2][64 * 40];
  __shared__ __align__(16) u16 Bs[2][64 * 40];
  const int tid = threadIdx.x, lane = tid & 31, w = tid >> 5;
  const int wm = (w >> 1) * 32, wn = (w & 1) * 32;
  const int m0 = (int)blockIdx.x * 64;
  const int n0 = (int)blockIdx.y * 64;
  const int n = lane & 15, hi = lane >> 4;

  auto load_tile = [&](int k0, int buf) {
#pragma unroll
    for (int i = 0; i < 2; ++i) {
      int chunk = tid * 2 + i;
      int r = chunk >> 2, cc = chunk & 3;
      *(u32x4*)&As[buf][r * 40 + cc * 8] =
          *(const u32x4*)&Y[(m0 + r) * CC + k0 + cc * 8];
      *(u32x4*)&Bs[buf][r * 40 + cc * 8] =
          *(const u32x4*)&wp[(n0 + r) * CC + k0 + cc * 8];
    }
  };

  v8f acc[2][2] = {};
  load_tile(0, 0);
  __syncthreads();
  int cur = 0;
  for (int k0 = 0; k0 < CC; k0 += 32) {
    if (k0 + 32 < CC) load_tile(k0 + 32, cur ^ 1);
    v16bf a[2], b[2];
#pragma unroll
    for (int i = 0; i < 2; ++i)
      a[i] = loadA(&As[cur][(wm + i * 16 + n) * 40], lane);
#pragma unroll
    for (int j = 0; j < 2; ++j)
      b[j] = loadB(&Bs[cur][(wn + j * 16 + n) * 40], lane);
#pragma unroll
    for (int i = 0; i < 2; ++i)
#pragma unroll
      for (int j = 0; j < 2; ++j) acc[i][j] = wmma_bf16(a[i], b[j], acc[i][j]);
    __syncthreads();
    cur ^= 1;
  }

#pragma unroll
  for (int i = 0; i < 2; ++i)
#pragma unroll
    for (int j = 0; j < 2; ++j) {
      int col = n0 + wn + j * 16 + n;
      float bv = bias[col];
      int base = (m0 + wm + i * 16 + hi * 8) * CC + col;
#pragma unroll
      for (int v = 0; v < 8; ++v)
        out[base + v * CC] = acc[i][j][v] + bv;
    }
}

// ---------------------------------------------------------------------------
// Launcher.
// ---------------------------------------------------------------------------
extern "C" void kernel_launch(void* const* d_in, const int* in_sizes, int n_in,
                              void* d_out, int out_size, void* d_ws,
                              size_t ws_size, hipStream_t stream) {
  const float* x      = (const float*)d_in[0];
  const float* qkv_w  = (const float*)d_in[1];
  const float* proj_w = (const float*)d_in[2];
  const float* proj_b = (const float*)d_in[3];
  (void)in_sizes; (void)n_in; (void)out_size; (void)ws_size;

  char* ws = (char*)d_ws;
  size_t off = 0;
  auto carve = [&](size_t bytes) -> void* {
    void* p = ws + off;
    off += (bytes + 255) & ~(size_t)255;
    return p;
  };
  // Workspace layout (~79.5 MB total).
  u16* xbf = (u16*)carve((size_t)MT * CC * 2);
  u16* wq  = (u16*)carve((size_t)C3 * CC * 2);
  u16* wp  = (u16*)carve((size_t)CC * CC * 2);
  u16* Q   = (u16*)carve((size_t)BB * HH * NN * HD * 2);
  u16* K   = (u16*)carve((size_t)BB * HH * NN * HD * 2);
  u16* Vt  = (u16*)carve((size_t)BB * HH * HD * NN * 2);
  u16* Y   = (u16*)carve((size_t)MT * CC * 2);

  {
    int n1 = MT * CC, n2 = C3 * CC, n3 = CC * CC;
    k_cvt<<<(n1 + 255) / 256, 256, 0, stream>>>(x, xbf, n1);
    k_cvt<<<(n2 + 255) / 256, 256, 0, stream>>>(qkv_w, wq, n2);
    k_cvt<<<(n3 + 255) / 256, 256, 0, stream>>>(proj_w, wp, n3);
  }
  k_qkv<<<dim3(MT / 64, C3 / 64), 128, 0, stream>>>(xbf, wq, Q, K, Vt);
  k_attn_t<<<BB * HH * (NT / 64),        128, 0, stream>>>(Q, K, Vt, Y);
  k_attn_s<<<BB * HH * ((NN - NT) / 64), 128, 0, stream>>>(Q, K, Vt, Y);
  k_proj<<<dim3(MT / 64, CC / 64), 128, 0, stream>>>(Y, wp, proj_b,
                                                     (float*)d_out);
}
